// PointNetTransformerSegFast_60507499266132
// MI455X (gfx1250) — matrix operand order repaired and checked by reading, not compiled, not run on hardware
//
#include <hip/hip_runtime.h>
#include <math.h>

typedef __attribute__((ext_vector_type(16))) _Float16 v16h;
typedef __attribute__((ext_vector_type(8)))  float    v8f;

#define Bx 2
#define Nx 16384
#define Mx 2048
#define Dx 256
#define NHEADx 8
#define DHx 32
#define FFx 512
#define Cx 50
#define HIDx 128
#define Lx 4

static __device__ __forceinline__ v8f wmma16(v16h a, v16h b, v8f c) {
  return __builtin_amdgcn_wmma_f32_16x16x32_f16(false, a, false, b, (short)0, c, false, false);
}

// ---------------------------------------------------------------- convert f32 -> f16
__global__ void k_cvt(const float* __restrict__ a, _Float16* __restrict__ o, int n) {
  int i = blockIdx.x * 256 + threadIdx.x;
  if (i < n) o[i] = (_Float16)a[i];
}

// ---------------------------------------------------------------- token embed: x = MLP_e(tok)+MLP_p(tok)
__global__ void k_embed(const float* __restrict__ xyz,
                        const float* __restrict__ ew1, const float* __restrict__ eb1,
                        const float* __restrict__ ew2, const float* __restrict__ eb2,
                        const float* __restrict__ pw1, const float* __restrict__ pb1,
                        const float* __restrict__ pw2, const float* __restrict__ pb2,
                        float* __restrict__ x) {
  __shared__ float hid[2 * HIDx];
  __shared__ float tk[3];
  int bm = blockIdx.x;                  // 0..4095 = b*M + m
  int b = bm >> 11, m = bm & (Mx - 1);
  int tid = threadIdx.x;
  if (tid < 3) tk[tid] = xyz[((size_t)b * Nx + (size_t)m * 8) * 3 + tid];
  __syncthreads();
  float t0 = tk[0], t1 = tk[1], t2 = tk[2];
  if (tid < HIDx) {
    float h = t0 * ew1[tid] + t1 * ew1[HIDx + tid] + t2 * ew1[2 * HIDx + tid] + eb1[tid];
    hid[tid] = fmaxf(h, 0.f);
  } else {
    int hh = tid - HIDx;
    float h = t0 * pw1[hh] + t1 * pw1[HIDx + hh] + t2 * pw1[2 * HIDx + hh] + pb1[hh];
    hid[HIDx + hh] = fmaxf(h, 0.f);
  }
  __syncthreads();
  float acc = eb2[tid] + pb2[tid];
  for (int h = 0; h < HIDx; ++h)
    acc += hid[h] * ew2[h * Dx + tid] + hid[HIDx + h] * pw2[h * Dx + tid];
  x[(size_t)bm * Dx + tid] = acc;
}

// ---------------------------------------------------------------- LayerNorm (one wave per 256-row), f16 out
__global__ void k_ln(const float* __restrict__ x, const float* __restrict__ s,
                     const float* __restrict__ bb, _Float16* __restrict__ y) {
  int wid = threadIdx.x >> 5, lane = threadIdx.x & 31;
  int row = blockIdx.x * 8 + wid;
  const float* xr = x + (size_t)row * Dx;
  float v[8]; float sum = 0.f;
#pragma unroll
  for (int j = 0; j < 8; ++j) { v[j] = xr[lane + 32 * j]; sum += v[j]; }
#pragma unroll
  for (int msk = 16; msk >= 1; msk >>= 1) sum += __shfl_xor(sum, msk, 32);
  float mean = sum * (1.f / 256.f);
  float vs = 0.f;
#pragma unroll
  for (int j = 0; j < 8; ++j) { float d = v[j] - mean; vs += d * d; }
#pragma unroll
  for (int msk = 16; msk >= 1; msk >>= 1) vs += __shfl_xor(vs, msk, 32);
  float inv = rsqrtf(vs * (1.f / 256.f) + 1e-5f);
#pragma unroll
  for (int j = 0; j < 8; ++j) {
    int c = lane + 32 * j;
    y[(size_t)row * Dx + c] = (_Float16)((v[j] - mean) * inv * s[c] + bb[c]);
  }
}

// ---------------------------------------------------------------- generic WMMA GEMM
// Block tile 128x64, 8 waves in 4x2 grid, wave tile 32x32 => 4 WMMA per K-step
// C(Mr x N) = act(A(Mr x K) f16 @ Bw(K x N) f16 + bias) [OMODE: 0=f16 out, 1=f32 out, 2=add into f32]
template <int ACT, int OMODE>
__global__ void k_gemm(const _Float16* __restrict__ A, const _Float16* __restrict__ Bw,
                       const float* __restrict__ bias, float* __restrict__ Cf,
                       _Float16* __restrict__ Ch, int Mr, int N, int K) {
  __shared__ __attribute__((aligned(16))) _Float16 sA[128 * 40];     // [row][k], pad 8
  __shared__ __attribute__((aligned(16))) _Float16 sBt[64 * 40];     // transposed [n][k]
  int tid = threadIdx.x;
  int wid = tid >> 5, lane = tid & 31;
  int lr = lane & 15, hs = lane >> 4;
  int wr = wid & 3, wc = wid >> 2;                                   // 4x2 wave grid
  int m0 = blockIdx.x * 128;
  int n0 = blockIdx.y * 64;
  v8f c00 = {}, c01 = {}, c10 = {}, c11 = {};
  int arow = tid >> 1, aseg = tid & 1;                               // 128 rows, 2x 16-half segs
  for (int k0 = 0; k0 < K; k0 += 32) {
    // stage A tile 128x32 (vector b128 loads, 2 per thread)
    const _Float16* ag = A + (size_t)(m0 + arow) * K + k0 + aseg * 16;
    uint4 av0 = *(const uint4*)(ag);
    uint4 av1 = *(const uint4*)(ag + 8);
    *(uint4*)(sA + arow * 40 + aseg * 16)     = av0;
    *(uint4*)(sA + arow * 40 + aseg * 16 + 8) = av1;
    if (k0 + 32 < K) __builtin_prefetch(ag + 32, 0, 0);              // -> global_prefetch_b8
    // stage B tile 32x64 transposed (scalar, N-guarded; N may be 50)
#pragma unroll
    for (int j = 0; j < 8; ++j) {
      int e = tid + 256 * j;
      int kr = e >> 6, nc = e & 63;
      _Float16 bv = (_Float16)0.f;
      if (n0 + nc < N) bv = Bw[(size_t)(k0 + kr) * N + n0 + nc];
      sBt[nc * 40 + kr] = bv;
    }
    __syncthreads();
    v16h a0, a1, b0, b1;
    const _Float16* ap0 = sA + (wr * 32 + lr) * 40 + 8 * hs;         // A-frag lane layout
    ((uint4*)&a0)[0] = *(const uint4*)(ap0);
    ((uint4*)&a0)[1] = *(const uint4*)(ap0 + 16);
    const _Float16* ap1 = sA + (wr * 32 + 16 + lr) * 40 + 8 * hs;
    ((uint4*)&a1)[0] = *(const uint4*)(ap1);
    ((uint4*)&a1)[1] = *(const uint4*)(ap1 + 16);
    const _Float16* bp0 = sBt + (wc * 32 + lr) * 40 + 8 * hs;        // B-frag lane layout
    ((uint4*)&b0)[0] = *(const uint4*)(bp0);
    ((uint4*)&b0)[1] = *(const uint4*)(bp0 + 16);
    const _Float16* bp1 = sBt + (wc * 32 + 16 + lr) * 40 + 8 * hs;
    ((uint4*)&b1)[0] = *(const uint4*)(bp1);
    ((uint4*)&b1)[1] = *(const uint4*)(bp1 + 16);
    c00 = wmma16(a0, b0, c00);
    c01 = wmma16(a0, b1, c01);
    c10 = wmma16(a1, b0, c10);
    c11 = wmma16(a1, b1, c11);
    __syncthreads();
  }
  int col0 = n0 + wc * 32 + lr;
  int col1 = col0 + 16;
  float bv0 = (col0 < N) ? bias[col0] : 0.f;
  float bv1 = (col1 < N) ? bias[col1] : 0.f;
#pragma unroll
  for (int r = 0; r < 8; ++r) {
    int row0 = m0 + wr * 32 + r + 8 * hs;
    int row1 = row0 + 16;
    float t00 = c00[r] + bv0, t01 = c01[r] + bv1;
    float t10 = c10[r] + bv0, t11 = c11[r] + bv1;
    if (ACT == 1) {
      t00 = fmaxf(t00, 0.f); t01 = fmaxf(t01, 0.f);
      t10 = fmaxf(t10, 0.f); t11 = fmaxf(t11, 0.f);
    }
    if (ACT == 2) {
      t00 = 0.5f * t00 * (1.f + erff(t00 * 0.70710678118654752f));
      t01 = 0.5f * t01 * (1.f + erff(t01 * 0.70710678118654752f));
      t10 = 0.5f * t10 * (1.f + erff(t10 * 0.70710678118654752f));
      t11 = 0.5f * t11 * (1.f + erff(t11 * 0.70710678118654752f));
    }
    if (col0 < N) {
      if (OMODE == 0)      { Ch[(size_t)row0 * N + col0] = (_Float16)t00; Ch[(size_t)row1 * N + col0] = (_Float16)t10; }
      else if (OMODE == 1) { Cf[(size_t)row0 * N + col0] = t00;           Cf[(size_t)row1 * N + col0] = t10; }
      else                 { Cf[(size_t)row0 * N + col0] += t00;          Cf[(size_t)row1 * N + col0] += t10; }
    }
    if (col1 < N) {
      if (OMODE == 0)      { Ch[(size_t)row0 * N + col1] = (_Float16)t01; Ch[(size_t)row1 * N + col1] = (_Float16)t11; }
      else if (OMODE == 1) { Cf[(size_t)row0 * N + col1] = t01;           Cf[(size_t)row1 * N + col1] = t11; }
      else                 { Cf[(size_t)row0 * N + col1] += t01;          Cf[(size_t)row1 * N + col1] += t11; }
    }
  }
}

// ---------------------------------------------------------------- flash attention (wave = 16-query tile of one head)
__global__ void k_flash(const _Float16* __restrict__ Q, const _Float16* __restrict__ Kt,
                        const _Float16* __restrict__ V, _Float16* __restrict__ O) {
  __shared__ __attribute__((aligned(16))) _Float16 sP[8][16 * 40];    // per-wave P tile
  __shared__ __attribute__((aligned(16))) _Float16 sVt[32 * 40];      // V chunk transposed [col][key]
  int tid = threadIdx.x;
  int wid = tid >> 5, lane = tid & 31, lr = lane & 15, hs = lane >> 4;
  int gw = blockIdx.x * 8 + wid;                // 0..2047
  int qt = gw & 127;
  int h  = (gw >> 7) & 7;                       // uniform within block (8 waves / 128 tiles-per-head)
  int b  = gw >> 10;
  const float scale = 0.17677669529663687f;     // 1/sqrt(32)
  size_t qbase = ((size_t)(b * Mx + qt * 16 + lr)) * Dx + h * DHx;
  v16h qf;
  ((uint4*)&qf)[0] = *(const uint4*)(Q + qbase + 8 * hs);
  ((uint4*)&qf)[1] = *(const uint4*)(Q + qbase + 16 + 8 * hs);
  float rm[8], rl[8];
  v8f o0 = {}, o1 = {};
#pragma unroll
  for (int r = 0; r < 8; ++r) { rm[r] = -3.0e38f; rl[r] = 0.f; }
  _Float16* myP = sP[wid];
  int vrow = tid >> 3, vseg = tid & 7;
  for (int c0 = 0; c0 < Mx; c0 += 32) {
    // cooperative stage of V chunk (32 keys x 32 dcols), transposed in LDS
    {
      const _Float16* vg = V + ((size_t)(b * Mx + c0 + vrow)) * Dx + h * DHx + vseg * 4;
      uint2 gv = *(const uint2*)(vg);
      if (c0 + 32 < Mx) __builtin_prefetch(vg + 32 * Dx, 0, 0);       // next chunk
      _Float16 tmp4[4];
      *(uint2*)tmp4 = gv;
#pragma unroll
      for (int j = 0; j < 4; ++j) sVt[(vseg * 4 + j) * 40 + vrow] = tmp4[j];
    }
    // S = Q @ K^T for two 16-key sub-tiles
    v16h kf0, kf1;
    size_t kb0 = ((size_t)(b * Mx + c0 + lr)) * Dx + h * DHx + hs * 8;
    size_t kb1 = ((size_t)(b * Mx + c0 + 16 + lr)) * Dx + h * DHx + hs * 8;
    ((uint4*)&kf0)[0] = *(const uint4*)(Kt + kb0);
    ((uint4*)&kf0)[1] = *(const uint4*)(Kt + kb0 + 16);
    ((uint4*)&kf1)[0] = *(const uint4*)(Kt + kb1);
    ((uint4*)&kf1)[1] = *(const uint4*)(Kt + kb1 + 16);
    v8f z0 = {}, z1 = {};
    v8f s0 = wmma16(qf, kf0, z0);
    v8f s1 = wmma16(qf, kf1, z1);
    // online softmax (row = r + 8*hs, cols across lr lanes)
#pragma unroll
    for (int r = 0; r < 8; ++r) {
      float a0 = s0[r] * scale, a1 = s1[r] * scale;
      float mx = fmaxf(a0, a1);
#pragma unroll
      for (int msk = 1; msk < 16; msk <<= 1) mx = fmaxf(mx, __shfl_xor(mx, msk, 32));
      float mnew = fmaxf(rm[r], mx);
      float alpha = __expf(rm[r] - mnew);
      float p0 = __expf(a0 - mnew), p1 = __expf(a1 - mnew);
      float ps = p0 + p1;
#pragma unroll
      for (int msk = 1; msk < 16; msk <<= 1) ps += __shfl_xor(ps, msk, 32);
      rl[r] = rl[r] * alpha + ps;
      rm[r] = mnew;
      o0[r] *= alpha; o1[r] *= alpha;
      myP[(r + 8 * hs) * 40 + lr]      = (_Float16)p0;
      myP[(r + 8 * hs) * 40 + 16 + lr] = (_Float16)p1;
    }
    __syncthreads();
    // P (A-frag) and V (B-frag) from LDS, accumulate O
    v16h pf, vf0, vf1;
    const _Float16* pp = myP + lr * 40 + 8 * hs;
    ((uint4*)&pf)[0] = *(const uint4*)(pp);
    ((uint4*)&pf)[1] = *(const uint4*)(pp + 16);
    const _Float16* vp0 = sVt + lr * 40 + 8 * hs;
    ((uint4*)&vf0)[0] = *(const uint4*)(vp0);
    ((uint4*)&vf0)[1] = *(const uint4*)(vp0 + 16);
    const _Float16* vp1 = sVt + (16 + lr) * 40 + 8 * hs;
    ((uint4*)&vf1)[0] = *(const uint4*)(vp1);
    ((uint4*)&vf1)[1] = *(const uint4*)(vp1 + 16);
    o0 = wmma16(pf, vf0, o0);
    o1 = wmma16(pf, vf1, o1);
    __syncthreads();
  }
#pragma unroll
  for (int r = 0; r < 8; ++r) {
    float inv = 1.f / rl[r];
    size_t orow = ((size_t)(b * Mx + qt * 16 + r + 8 * hs)) * Dx + h * DHx;
    O[orow + lr]      = (_Float16)(o0[r] * inv);
    O[orow + 16 + lr] = (_Float16)(o1[r] * inv);
  }
}

// ---------------------------------------------------------------- KNN top-3 + gather/mean
__global__ void k_knn(const float* __restrict__ xyz, const float* __restrict__ logits,
                      float* __restrict__ out) {
  __shared__ float tx[Mx], ty[Mx], tz[Mx], t2[Mx];
  int b = blockIdx.x >> 6;          // 64 blocks per batch
  int nb = blockIdx.x & 63;
  int tid = threadIdx.x;
  for (int m = tid; m < Mx; m += 256) {
    const float* p = xyz + ((size_t)b * Nx + (size_t)m * 8) * 3;
    float a = p[0], c = p[1], d = p[2];
    tx[m] = a; ty[m] = c; tz[m] = d; t2[m] = a * a + c * c + d * d;
  }
  __syncthreads();
  int n = nb * 256 + tid;
  const float* q = xyz + ((size_t)b * Nx + n) * 3;
  float qx = q[0], qy = q[1], qz = q[2];
  float q2 = qx * qx + qy * qy + qz * qz;
  float d0 = 3.4e38f, d1 = 3.4e38f, d2 = 3.4e38f;
  int i0 = 0, i1 = 0, i2 = 0;
  for (int m = 0; m < Mx; ++m) {
    float dist = q2 + t2[m] - 2.f * (qx * tx[m] + qy * ty[m] + qz * tz[m]);
    dist = fmaxf(dist, 0.f);
    if (dist < d2) {
      if (dist < d0)      { d2 = d1; i2 = i1; d1 = d0; i1 = i0; d0 = dist; i0 = m; }
      else if (dist < d1) { d2 = d1; i2 = i1; d1 = dist; i1 = m; }
      else                { d2 = dist; i2 = m; }
    }
  }
  const float* L0 = logits + ((size_t)b * Mx + i0) * Cx;
  const float* L1 = logits + ((size_t)b * Mx + i1) * Cx;
  const float* L2 = logits + ((size_t)b * Mx + i2) * Cx;
  float* o = out + ((size_t)b * Nx + n) * Cx;
  for (int c = 0; c < Cx; ++c) o[c] = (L0[c] + L1[c] + L2[c]) * (1.f / 3.f);
}

// ================================================================ host orchestration
extern "C" void kernel_launch(void* const* d_in, const int* in_sizes, int n_in,
                              void* d_out, int out_size, void* d_ws, size_t ws_size,
                              hipStream_t stream) {
  (void)in_sizes; (void)n_in; (void)out_size; (void)ws_size;
  const float* xyz = (const float*)d_in[0];
  const float* ew1 = (const float*)d_in[1];  const float* eb1 = (const float*)d_in[2];
  const float* ew2 = (const float*)d_in[3];  const float* eb2 = (const float*)d_in[4];
  const float* pw1 = (const float*)d_in[5];  const float* pb1 = (const float*)d_in[6];
  const float* pw2 = (const float*)d_in[7];  const float* pb2 = (const float*)d_in[8];
  const float* Wq = (const float*)d_in[9];   const float* bq = (const float*)d_in[10];
  const float* Wk = (const float*)d_in[11];  const float* bk = (const float*)d_in[12];
  const float* Wv = (const float*)d_in[13];  const float* bv = (const float*)d_in[14];
  const float* Wo = (const float*)d_in[15];  const float* bo = (const float*)d_in[16];
  const float* l1s = (const float*)d_in[17]; const float* l1b = (const float*)d_in[18];
  const float* W1 = (const float*)d_in[19];  const float* b1 = (const float*)d_in[20];
  const float* W2 = (const float*)d_in[21];  const float* b2 = (const float*)d_in[22];
  const float* l2s = (const float*)d_in[23]; const float* l2b = (const float*)d_in[24];
  const float* hw1 = (const float*)d_in[25]; const float* hb1 = (const float*)d_in[26];
  const float* hw2 = (const float*)d_in[27]; const float* hb2 = (const float*)d_in[28];

  const int R = Bx * Mx;                 // 4096 token rows
  char* w = (char*)d_ws;
  size_t off = 0;
  auto alloc = [&](size_t bytes) { size_t o = off; off += (bytes + 255) & ~(size_t)255; return o; };
  float*     x    = (float*)(w + alloc((size_t)R * Dx * 4));
  _Float16*  y16  = (_Float16*)(w + alloc((size_t)R * Dx * 2));
  _Float16*  q16  = (_Float16*)(w + alloc((size_t)R * Dx * 2));
  _Float16*  k16  = (_Float16*)(w + alloc((size_t)R * Dx * 2));
  _Float16*  v16  = (_Float16*)(w + alloc((size_t)R * Dx * 2));
  _Float16*  o16  = (_Float16*)(w + alloc((size_t)R * Dx * 2));
  _Float16*  h16  = (_Float16*)(w + alloc((size_t)R * FFx * 2));
  float*     ltok = (float*)(w + alloc((size_t)R * Cx * 4));
  _Float16*  Wq16 = (_Float16*)(w + alloc((size_t)Lx * Dx * Dx * 2));
  _Float16*  Wk16 = (_Float16*)(w + alloc((size_t)Lx * Dx * Dx * 2));
  _Float16*  Wv16 = (_Float16*)(w + alloc((size_t)Lx * Dx * Dx * 2));
  _Float16*  Wo16 = (_Float16*)(w + alloc((size_t)Lx * Dx * Dx * 2));
  _Float16*  W116 = (_Float16*)(w + alloc((size_t)Lx * Dx * FFx * 2));
  _Float16*  W216 = (_Float16*)(w + alloc((size_t)Lx * FFx * Dx * 2));
  _Float16*  hw116 = (_Float16*)(w + alloc((size_t)Dx * 256 * 2));
  _Float16*  hw216 = (_Float16*)(w + alloc((size_t)256 * Cx * 2));

  auto cvt = [&](const float* src, _Float16* dst, int n) {
    k_cvt<<<(n + 255) / 256, 256, 0, stream>>>(src, dst, n);
  };
  cvt(Wq, Wq16, Lx * Dx * Dx);  cvt(Wk, Wk16, Lx * Dx * Dx);
  cvt(Wv, Wv16, Lx * Dx * Dx);  cvt(Wo, Wo16, Lx * Dx * Dx);
  cvt(W1, W116, Lx * Dx * FFx); cvt(W2, W216, Lx * FFx * Dx);
  cvt(hw1, hw116, Dx * 256);    cvt(hw2, hw216, 256 * Cx);

  k_embed<<<R, 256, 0, stream>>>(xyz, ew1, eb1, ew2, eb2, pw1, pb1, pw2, pb2, x);

  dim3 gD(R / 128, Dx / 64);     // N=256 -> (32,4)
  dim3 gF(R / 128, FFx / 64);    // N=512 -> (32,8)
  dim3 gC(R / 128, (Cx + 63) / 64);
  for (int l = 0; l < Lx; ++l) {
    k_ln<<<R / 8, 256, 0, stream>>>(x, l1s + l * Dx, l1b + l * Dx, y16);
    k_gemm<0, 0><<<gD, 256, 0, stream>>>(y16, Wq16 + (size_t)l * Dx * Dx, bq + l * Dx, nullptr, q16, R, Dx, Dx);
    k_gemm<0, 0><<<gD, 256, 0, stream>>>(y16, Wk16 + (size_t)l * Dx * Dx, bk + l * Dx, nullptr, k16, R, Dx, Dx);
    k_gemm<0, 0><<<gD, 256, 0, stream>>>(y16, Wv16 + (size_t)l * Dx * Dx, bv + l * Dx, nullptr, v16, R, Dx, Dx);
    k_flash<<<256, 256, 0, stream>>>(q16, k16, v16, o16);
    k_gemm<0, 2><<<gD, 256, 0, stream>>>(o16, Wo16 + (size_t)l * Dx * Dx, bo + l * Dx, x, nullptr, R, Dx, Dx);
    k_ln<<<R / 8, 256, 0, stream>>>(x, l2s + l * Dx, l2b + l * Dx, y16);
    k_gemm<2, 0><<<gF, 256, 0, stream>>>(y16, W116 + (size_t)l * Dx * FFx, b1 + l * FFx, nullptr, h16, R, FFx, Dx);
    k_gemm<0, 2><<<gD, 256, 0, stream>>>(h16, W216 + (size_t)l * FFx * Dx, b2 + l * Dx, x, nullptr, R, Dx, FFx);
  }

  // segmentation head on tokens
  cvt(x, y16, R * Dx);
  k_gemm<1, 0><<<gD, 256, 0, stream>>>(y16, hw116, hb1, nullptr, h16, R, 256, Dx);
  k_gemm<0, 1><<<gC, 256, 0, stream>>>(h16, hw216, hb2, ltok, nullptr, R, Cx, 256);

  // KNN interpolate to all N points
  k_knn<<<Bx * (Nx / 256), 256, 0, stream>>>(xyz, ltok, (float*)d_out);
}
